// CrossAttentionBlock_35777077575804
// MI455X (gfx1250) — matrix operand order repaired and checked by reading, not compiled, not run on hardware
//
#include <hip/hip_runtime.h>
#include <hip/hip_bf16.h>
#include <stdint.h>

// ---------------------------------------------------------------------------
// CDNA5 (gfx1250) CrossAttentionBlock, bf16 WMMA pipeline.
// Workspace layout (bytes), total ~101 MB:
//   wq_bf16   [1024x1024]  @ 0
//   wkv_bf16  [2048x1024]  @ 2 MiB
//   wo_bf16   [1024x1024]  @ 6 MiB
//   s_x f32   [16x1024]    @ 8 MiB
//   s_c f32   [16x1024]    @ 8 MiB + 64 KiB
//   Aq bf16   [16384x1024] @ ~8.13 MiB   (rms-normed x; REUSED later as O)
//   Akv bf16  [4096x1024]
//   q  bf16   [16384x1024]
//   kv bf16   [4096x2048]
// ---------------------------------------------------------------------------

typedef __attribute__((ext_vector_type(16))) __bf16 v16bf;
typedef __attribute__((ext_vector_type(8)))  float  v8f;
typedef __attribute__((ext_vector_type(4)))  int    v4i;

union Frag { v16bf v; uint4 q[2]; };

// ---- async global->LDS path (gfx1250 GLOBAL_LOAD_ASYNC_TO_LDS_B128) -------
#if defined(__has_builtin)
#if __has_builtin(__builtin_amdgcn_global_load_async_to_lds_b128)
#define HAVE_ASYNC_LDS 1
#endif
#endif

__device__ __forceinline__ void async_copy16(void* dst_lds, const void* src_glob) {
#ifdef HAVE_ASYNC_LDS
    __builtin_amdgcn_global_load_async_to_lds_b128(
        (__attribute__((address_space(1))) v4i*)(src_glob),
        (__attribute__((address_space(3))) v4i*)(dst_lds), 0, 0);
#else
    *(uint4*)dst_lds = *(const uint4*)src_glob;
#endif
}

__device__ __forceinline__ void wait_async() {
#ifdef HAVE_ASYNC_LDS
#if __has_builtin(__builtin_amdgcn_s_wait_asynccnt)
    __builtin_amdgcn_s_wait_asynccnt(0);
#else
    asm volatile("s_wait_asynccnt 0" ::: "memory");
#endif
#endif
}

__device__ __forceinline__ float bf2f(uint32_t h) {
    return __uint_as_float(h << 16);
}
__device__ __forceinline__ uint16_t f2bf(float f) {
    uint32_t u = __float_as_uint(f);
    uint32_t r = u + 0x7FFFu + ((u >> 16) & 1u);   // round-to-nearest-even
    return (uint16_t)(r >> 16);
}
__device__ __forceinline__ void unpack8(uint4 v, float* f) {
    uint32_t w[4] = {v.x, v.y, v.z, v.w};
#pragma unroll
    for (int i = 0; i < 4; ++i) {
        f[2 * i]     = bf2f(w[i] & 0xFFFFu);
        f[2 * i + 1] = bf2f(w[i] >> 16);
    }
}
__device__ __forceinline__ uint4 pack8(const float* f) {
    uint32_t w[4];
#pragma unroll
    for (int i = 0; i < 4; ++i)
        w[i] = (uint32_t)f2bf(f[2 * i]) | ((uint32_t)f2bf(f[2 * i + 1]) << 16);
    uint4 v; v.x = w[0]; v.y = w[1]; v.z = w[2]; v.w = w[3];
    return v;
}

// ------------------------------- conversion --------------------------------
__global__ __launch_bounds__(256) void k_f32_to_bf16(const float* __restrict__ in,
                                                     uint16_t* __restrict__ out, int n) {
    int i = blockIdx.x * 256 + threadIdx.x;
    if (i < n) out[i] = f2bf(in[i]);
}

// ---------------- s = cond @ W^T + 1   (cond [16,768], W [dout,768]) -------
__global__ __launch_bounds__(256) void k_cond_scale(const float* __restrict__ cond,
                                                    const float* __restrict__ W,
                                                    float* __restrict__ S,
                                                    int dout, int df) {
    int gid = blockIdx.x * 256 + threadIdx.x;
    if (gid >= 16 * dout) return;
    int b = gid / dout, j = gid % dout;
    const float4* c4 = (const float4*)(cond + (size_t)b * df);
    const float4* w4 = (const float4*)(W + (size_t)j * df);
    float acc = 0.f;
    for (int f = 0; f < df / 4; ++f) {
        float4 a = c4[f], w = w4[f];
        acc += a.x * w.x + a.y * w.y + a.z * w.z + a.w * w.w;
    }
    S[gid] = acc + 1.0f;
}

// ---- AdaRMSNorm row (1024 cols): out_bf16 = x * S[batch,c] * rsqrt(mean+eps)
__global__ __launch_bounds__(256) void k_rmsnorm_bf16(const float* __restrict__ X,
                                                      const float* __restrict__ S,
                                                      uint16_t* __restrict__ out,
                                                      int rows_per_batch) {
    __shared__ float red[256];
    int row = blockIdx.x, tid = threadIdx.x;
    size_t base = (size_t)row * 1024 + tid * 4;
    float4 v = *(const float4*)(X + base);
    float p = v.x * v.x + v.y * v.y + v.z * v.z + v.w * v.w;
    red[tid] = p;
    __syncthreads();
#pragma unroll
    for (int s = 128; s > 0; s >>= 1) {
        if (tid < s) red[tid] += red[tid + s];
        __syncthreads();
    }
    float ms = red[0] * (1.0f / 1024.0f);
    float r = rsqrtf(ms + 1e-6f);
    int batch = row / rows_per_batch;
    float4 sc = *(const float4*)(S + (size_t)batch * 1024 + tid * 4);
    float f[4] = {v.x * sc.x * r, v.y * sc.y * r, v.z * sc.z * r, v.w * sc.w * r};
    uint32_t w0 = (uint32_t)f2bf(f[0]) | ((uint32_t)f2bf(f[1]) << 16);
    uint32_t w1 = (uint32_t)f2bf(f[2]) | ((uint32_t)f2bf(f[3]) << 16);
    uint2 pk; pk.x = w0; pk.y = w1;
    *(uint2*)(out + base) = pk;
}

// -------------------- bf16 WMMA GEMM: C[M,N] = A[M,K] @ B[N,K]^T -----------
// Block 128x128, BK=32, 256 threads = 8 waves in 2(M)x4(N); wave tile 64x32.
// Double-buffered K loop: async global->LDS copies for the next tile are
// issued right after the barrier, overlapping with WMMA on the current tile.
// One barrier per iteration: the buffer being refilled was last *read* in the
// previous iteration, and every wave passed this barrier after that read.
template <bool BF16_OUT, bool ADD_SKIP>
__global__ __launch_bounds__(256) void k_gemm_bf16(const uint16_t* __restrict__ A,
                                                   const uint16_t* __restrict__ B,
                                                   float* __restrict__ Cf,
                                                   uint16_t* __restrict__ Cb,
                                                   const float* __restrict__ skip,
                                                   int M, int N, int K) {
    __shared__ uint16_t As[2][128][40];   // 32 + 8 pad, double buffered
    __shared__ uint16_t Bs[2][128][40];
    const int tid = threadIdx.x;
    const int wave = tid >> 5, lane = tid & 31, lr = lane & 15, lh = lane >> 4;
    const int wm = (wave >> 2) * 64, wn = (wave & 3) * 32;
    const int bm = blockIdx.y * 128, bn = blockIdx.x * 128;

    auto issue_tile = [&](int buf, int k0) {
#pragma unroll
        for (int i = 0; i < 2; ++i) {
            int c = tid + i * 256;          // 512 16B chunks per tile
            int row = c >> 2, cc = c & 3;
            async_copy16(&As[buf][row][cc * 8],
                         A + (size_t)(bm + row) * K + k0 + cc * 8);
            async_copy16(&Bs[buf][row][cc * 8],
                         B + (size_t)(bn + row) * K + k0 + cc * 8);
        }
    };

    v8f acc[4][2] = {};

    issue_tile(0, 0);
    for (int k0 = 0, it = 0; k0 < K; k0 += 32, ++it) {
        const int cur = it & 1;
        wait_async();
        __syncthreads();
        if (k0 + 32 < K) issue_tile(cur ^ 1, k0 + 32);

        Frag a[4];
#pragma unroll
        for (int m = 0; m < 4; ++m) {
            int r = wm + m * 16 + lr;
            // 16-bit A frag: lanes 0-15 hold K 0..7 & 16..23; lanes 16-31 hold 8..15 & 24..31
            a[m].q[0] = *(const uint4*)&As[cur][r][lh * 8];
            a[m].q[1] = *(const uint4*)&As[cur][r][lh * 8 + 16];
        }
#pragma unroll
        for (int nn = 0; nn < 2; ++nn) {
            Frag bf;
            int n = wn + nn * 16 + lr;
            // B frag: lanes 0-15 hold K 0..7 & 16..23; lanes 16-31 hold 8..15 & 24..31
            bf.q[0] = *(const uint4*)&Bs[cur][n][lh * 8];
            bf.q[1] = *(const uint4*)&Bs[cur][n][lh * 8 + 16];
#pragma unroll
            for (int m = 0; m < 4; ++m)
                acc[m][nn] = __builtin_amdgcn_wmma_f32_16x16x32_bf16(
                    false, a[m].v, false, bf.v, (short)0, acc[m][nn], false, false);
        }
    }

#pragma unroll
    for (int m = 0; m < 4; ++m)
#pragma unroll
        for (int nn = 0; nn < 2; ++nn)
#pragma unroll
            for (int j = 0; j < 8; ++j) {
                int row = bm + wm + m * 16 + j + lh * 8;
                int col = bn + wn + nn * 16 + lr;
                size_t idx = (size_t)row * N + col;
                float v = acc[m][nn][j];
                if (ADD_SKIP) v += skip[idx];
                if (BF16_OUT) Cb[idx] = f2bf(v);
                else          Cf[idx] = v;
            }
}

// ------------------------------- attention ---------------------------------
// grid = (qchunk 8, head 16, batch 16), 256 threads (8 waves x 16 q-rows).
// Dynamic LDS layout (bytes):
//   maskadd f32[256]      @ 0
//   qred    f32[256]      @ 1024
//   Vt  u16[64][264]      @ 2048      (V transposed: rows = head dim)
//   Ks  u16[256][72]      @ 35840
//   Qs  u16[128][72]      @ 72704
//   Ps  u16[128][264]     @ 35840     (aliases Ks+Qs after scores done)
// total = 103424 bytes (< 320 KB/WGP).
__global__ __launch_bounds__(256) void k_attention(const uint16_t* __restrict__ Qg,
                                                   const uint16_t* __restrict__ KVg,
                                                   const uint8_t* __restrict__ maskb,
                                                   const float* __restrict__ qk_scale,
                                                   uint16_t* __restrict__ Og) {
    extern __shared__ char smem[];
    float*    maskadd = (float*)smem;
    float*    qred    = (float*)(smem + 1024);
    uint16_t* Vt      = (uint16_t*)(smem + 2048);
    uint16_t* Ks      = (uint16_t*)(smem + 35840);
    uint16_t* Qs      = (uint16_t*)(smem + 72704);
    uint16_t* Ps      = (uint16_t*)(smem + 35840);

    const int tid = threadIdx.x;
    const int qc = blockIdx.x, head = blockIdx.y, b = blockIdx.z;
    const int wave = tid >> 5, lane = tid & 31, lr = lane & 15, lh = lane >> 4;
    const float sqsc = sqrtf(qk_scale[head]);

    maskadd[tid] = maskb[b * 256 + tid] ? 0.0f : -1e30f;

    // ---- stage K (cosine-normalized) and V (transposed) ----
    {
        int l = tid;   // key index 0..255
        size_t kbase = ((size_t)b * 256 + l) * 2048 + (size_t)head * 64;
        uint4 kr[8];
        float ss = 0.f;
#pragma unroll
        for (int c = 0; c < 8; ++c) {
            kr[c] = *(const uint4*)(KVg + kbase + c * 8);
            float f[8]; unpack8(kr[c], f);
#pragma unroll
            for (int e = 0; e < 8; ++e) ss += f[e] * f[e];
        }
        float ksc = sqsc * rsqrtf(ss + 1e-6f);
#pragma unroll
        for (int c = 0; c < 8; ++c) {
            float f[8]; unpack8(kr[c], f);
#pragma unroll
            for (int e = 0; e < 8; ++e) f[e] *= ksc;
            *(uint4*)&Ks[l * 72 + c * 8] = pack8(f);
        }
        size_t vbase = kbase + 1024;
#pragma unroll
        for (int c = 0; c < 8; ++c) {
            uint4 vv = *(const uint4*)(KVg + vbase + c * 8);
            uint32_t w[4] = {vv.x, vv.y, vv.z, vv.w};
#pragma unroll
            for (int i = 0; i < 4; ++i) {
                Vt[(c * 8 + 2 * i) * 264 + l]     = (uint16_t)(w[i] & 0xFFFFu);
                Vt[(c * 8 + 2 * i + 1) * 264 + l] = (uint16_t)(w[i] >> 16);
            }
        }
    }

    // ---- stage Q (cosine-normalized, 1/sqrt(64) folded in) ----
    int qrw = tid & 127, half = tid >> 7;
    size_t qbase = ((size_t)b * 1024 + qc * 128 + qrw) * 1024 + (size_t)head * 64 + half * 32;
    uint4 qr[4];
    float qs = 0.f;
#pragma unroll
    for (int c = 0; c < 4; ++c) {
        qr[c] = *(const uint4*)(Qg + qbase + c * 8);
        float f[8]; unpack8(qr[c], f);
#pragma unroll
        for (int e = 0; e < 8; ++e) qs += f[e] * f[e];
    }
    qred[tid] = qs;
    __syncthreads();
    float qtot = qred[tid] + qred[tid ^ 128];
    float qscale = sqsc * rsqrtf(qtot + 1e-6f) * 0.125f;
#pragma unroll
    for (int c = 0; c < 4; ++c) {
        float f[8]; unpack8(qr[c], f);
#pragma unroll
        for (int e = 0; e < 8; ++e) f[e] *= qscale;
        *(uint4*)&Qs[qrw * 72 + half * 32 + c * 8] = pack8(f);
    }
    __syncthreads();

    // ---- scores: S[16 x 256] per wave via WMMA bf16 ----
    const int wr = wave * 16;
    const int qrow = wr + lr;
    Frag a0, a1;
    a0.q[0] = *(const uint4*)&Qs[qrow * 72 + lh * 8];
    a0.q[1] = *(const uint4*)&Qs[qrow * 72 + lh * 8 + 16];
    a1.q[0] = *(const uint4*)&Qs[qrow * 72 + 32 + lh * 8];
    a1.q[1] = *(const uint4*)&Qs[qrow * 72 + 32 + lh * 8 + 16];

    v8f sc[16];
#pragma unroll
    for (int t = 0; t < 16; ++t) {
        Frag b0, b1;
        int key = t * 16 + lr;
        b0.q[0] = *(const uint4*)&Ks[key * 72 + lh * 8];
        b0.q[1] = *(const uint4*)&Ks[key * 72 + lh * 8 + 16];
        b1.q[0] = *(const uint4*)&Ks[key * 72 + 32 + lh * 8];
        b1.q[1] = *(const uint4*)&Ks[key * 72 + 32 + lh * 8 + 16];
        v8f z = {};
        z = __builtin_amdgcn_wmma_f32_16x16x32_bf16(false, a0.v, false, b0.v, (short)0, z, false, false);
        sc[t] = __builtin_amdgcn_wmma_f32_16x16x32_bf16(false, a1.v, false, b1.v, (short)0, z, false, false);
    }
    __syncthreads();   // everyone done reading Ks/Qs before Ps overwrites them

    // ---- mask + softmax (fp32, half-wave shfl reductions; wave32) ----
#pragma unroll
    for (int t = 0; t < 16; ++t) {
        float ma = maskadd[t * 16 + lr];
#pragma unroll
        for (int j = 0; j < 8; ++j) sc[t][j] += ma;
    }
    float rinv[8];
#pragma unroll
    for (int j = 0; j < 8; ++j) {
        float m = -3.0e38f;
#pragma unroll
        for (int t = 0; t < 16; ++t) m = fmaxf(m, sc[t][j]);
#pragma unroll
        for (int off = 1; off < 16; off <<= 1) m = fmaxf(m, __shfl_xor(m, off, 32));
        float s = 0.f;
#pragma unroll
        for (int t = 0; t < 16; ++t) {
            float e = __expf(sc[t][j] - m);
            sc[t][j] = e;
            s += e;
        }
#pragma unroll
        for (int off = 1; off < 16; off <<= 1) s += __shfl_xor(s, off, 32);
        rinv[j] = 1.0f / s;
    }
#pragma unroll
    for (int t = 0; t < 16; ++t)
#pragma unroll
        for (int j = 0; j < 8; ++j)
            Ps[(wr + j + lh * 8) * 264 + t * 16 + lr] = f2bf(sc[t][j] * rinv[j]);
    __syncthreads();

    // ---- O = P @ V via WMMA (K = 256 in 8 steps of 32) ----
    v8f o[4] = {};
#pragma unroll
    for (int s = 0; s < 8; ++s) {
        Frag pa;
        pa.q[0] = *(const uint4*)&Ps[qrow * 264 + s * 32 + lh * 8];
        pa.q[1] = *(const uint4*)&Ps[qrow * 264 + s * 32 + lh * 8 + 16];
#pragma unroll
        for (int nn = 0; nn < 4; ++nn) {
            Frag vb;
            int n = nn * 16 + lr;
            vb.q[0] = *(const uint4*)&Vt[n * 264 + s * 32 + lh * 8];
            vb.q[1] = *(const uint4*)&Vt[n * 264 + s * 32 + lh * 8 + 16];
            o[nn] = __builtin_amdgcn_wmma_f32_16x16x32_bf16(false, pa.v, false, vb.v,
                                                            (short)0, o[nn], false, false);
        }
    }
#pragma unroll
    for (int nn = 0; nn < 4; ++nn)
#pragma unroll
        for (int j = 0; j < 8; ++j) {
            size_t grow = (size_t)b * 1024 + qc * 128 + wr + j + lh * 8;
            size_t gcol = (size_t)head * 64 + nn * 16 + lr;
            Og[grow * 1024 + gcol] = f2bf(o[nn][j]);
        }
}

// ---------------------------------------------------------------------------
extern "C" void kernel_launch(void* const* d_in, const int* in_sizes, int n_in,
                              void* d_out, int out_size, void* d_ws, size_t ws_size,
                              hipStream_t stream) {
    (void)in_sizes; (void)n_in; (void)out_size; (void)ws_size;
    const float*   x       = (const float*)d_in[0];   // [16,32,32,1024]
    const float*   cond    = (const float*)d_in[1];   // [16,768]
    const float*   ccond   = (const float*)d_in[2];   // [16,256,1024]
    const uint8_t* maskp   = (const uint8_t*)d_in[3]; // [16,256] bool
    const float*   w_norm  = (const float*)d_in[4];   // [1024,768]
    const float*   w_q     = (const float*)d_in[5];   // [1024,1024]
    const float*   w_cnorm = (const float*)d_in[6];   // [1024,768]
    const float*   w_kv    = (const float*)d_in[7];   // [2048,1024]
    const float*   qk_sc   = (const float*)d_in[8];   // [16]
    const float*   w_o     = (const float*)d_in[9];   // [1024,1024]
    float* out = (float*)d_out;

    char* ws = (char*)d_ws;
    const size_t off_wq  = 0;
    const size_t off_wkv = off_wq  + (size_t)1024 * 1024 * 2;
    const size_t off_wo  = off_wkv + (size_t)2048 * 1024 * 2;
    const size_t off_sx  = off_wo  + (size_t)1024 * 1024 * 2;
    const size_t off_sc  = off_sx  + (size_t)16 * 1024 * 4;
    const size_t off_Aq  = off_sc  + (size_t)16 * 1024 * 4;
    const size_t off_Akv = off_Aq  + (size_t)16384 * 1024 * 2;
    const size_t off_q   = off_Akv + (size_t)4096 * 1024 * 2;
    const size_t off_kv  = off_q   + (size_t)16384 * 1024 * 2;
    const size_t off_O   = off_Aq;   // reuse: Aq dead after q-projection

    uint16_t* wq_b  = (uint16_t*)(ws + off_wq);
    uint16_t* wkv_b = (uint16_t*)(ws + off_wkv);
    uint16_t* wo_b  = (uint16_t*)(ws + off_wo);
    float*    sx    = (float*)(ws + off_sx);
    float*    scnd  = (float*)(ws + off_sc);
    uint16_t* Aq    = (uint16_t*)(ws + off_Aq);
    uint16_t* Akv   = (uint16_t*)(ws + off_Akv);
    uint16_t* qb    = (uint16_t*)(ws + off_q);
    uint16_t* kvb   = (uint16_t*)(ws + off_kv);
    uint16_t* Ob    = (uint16_t*)(ws + off_O);

    // 1) weights -> bf16
    k_f32_to_bf16<<<(1024 * 1024) / 256, 256, 0, stream>>>(w_q,  wq_b,  1024 * 1024);
    k_f32_to_bf16<<<(2048 * 1024) / 256, 256, 0, stream>>>(w_kv, wkv_b, 2048 * 1024);
    k_f32_to_bf16<<<(1024 * 1024) / 256, 256, 0, stream>>>(w_o,  wo_b,  1024 * 1024);

    // 2) AdaRMSNorm scales
    k_cond_scale<<<(16 * 1024) / 256, 256, 0, stream>>>(cond, w_norm,  sx,   1024, 768);
    k_cond_scale<<<(16 * 1024) / 256, 256, 0, stream>>>(cond, w_cnorm, scnd, 1024, 768);

    // 3) RMS-norm activations -> bf16
    k_rmsnorm_bf16<<<16384, 256, 0, stream>>>(x,     sx,   Aq,  1024);
    k_rmsnorm_bf16<<<4096,  256, 0, stream>>>(ccond, scnd, Akv, 256);

    // 4) projections (bf16 WMMA GEMMs, async double-buffered)
    k_gemm_bf16<true, false><<<dim3(1024 / 128, 16384 / 128), 256, 0, stream>>>(
        Aq, wq_b, nullptr, qb, nullptr, 16384, 1024, 1024);
    k_gemm_bf16<true, false><<<dim3(2048 / 128, 4096 / 128), 256, 0, stream>>>(
        Akv, wkv_b, nullptr, kvb, nullptr, 4096, 2048, 1024);

    // 5) attention (cosine-sim norm fused, WMMA QK^T and PV)
    k_attention<<<dim3(8, 16, 16), 256, 103424, stream>>>(qb, kvb, maskp, qk_sc, Ob);

    // 6) output projection + residual (f32 out)
    k_gemm_bf16<false, true><<<dim3(1024 / 128, 16384 / 128), 256, 0, stream>>>(
        Ob, wo_b, out, nullptr, x, 16384, 1024, 1024);
}
